// TransformerBlock_8856222565048
// MI455X (gfx1250) — matrix operand order repaired and checked by reading, not compiled, not run on hardware
//
#include <hip/hip_runtime.h>

typedef __attribute__((ext_vector_type(16))) __bf16 v16bf;
typedef __attribute__((ext_vector_type(8)))  float  v8f;

union FragB16 { v16bf v; unsigned int u[8]; };

__device__ __forceinline__ unsigned short f2bf(float f) {
    union { float f; unsigned u; } x; x.f = f;
    unsigned r = x.u + 0x7fffu + ((x.u >> 16) & 1u);
    return (unsigned short)(r >> 16);
}

__global__ __launch_bounds__(256) void cvt_f32_bf16(const float* __restrict__ in,
                                                    unsigned short* __restrict__ out, int n) {
    int i = blockIdx.x * 256 + threadIdx.x;
    if (i < n) out[i] = f2bf(in[i]);
}

// LayerNorm over E=1024; one block (256 thr) per row; bf16 output.
__global__ __launch_bounds__(256) void ln_bf16(const float* __restrict__ x,
                                               const float* __restrict__ g,
                                               const float* __restrict__ b,
                                               unsigned short* __restrict__ out) {
    __shared__ float red[8], red2[8];
    const int row = blockIdx.x;
    const float* xr = x + (size_t)row * 1024;
    float v[4], s = 0.f, s2 = 0.f;
#pragma unroll
    for (int i = 0; i < 4; ++i) {
        v[i] = xr[threadIdx.x + i * 256];
        s += v[i]; s2 += v[i] * v[i];
    }
#pragma unroll
    for (int off = 16; off > 0; off >>= 1) {
        s  += __shfl_xor(s,  off);
        s2 += __shfl_xor(s2, off);
    }
    int wave = threadIdx.x >> 5, lane = threadIdx.x & 31;
    if (lane == 0) { red[wave] = s; red2[wave] = s2; }
    __syncthreads();
    float ts = 0.f, ts2 = 0.f;
#pragma unroll
    for (int i = 0; i < 8; ++i) { ts += red[i]; ts2 += red2[i]; }
    float mean = ts * (1.0f / 1024.0f);
    float var  = ts2 * (1.0f / 1024.0f) - mean * mean;
    float rs   = rsqrtf(var + 1e-5f);
    unsigned short* orow = out + (size_t)row * 1024;
#pragma unroll
    for (int i = 0; i < 4; ++i) {
        int c = threadIdx.x + i * 256;
        orow[c] = f2bf((v[i] - mean) * rs * g[c] + b[c]);
    }
}

// Tiled bf16 GEMM: C = A[MxK] * B[KxN] (+bias)(+gelu | +resid)
// Block tile 256x128, 8 waves in 4(M)x2(N) grid, each wave 64x64 (4x4 WMMA tiles).
// mode 1: bias -> bf16 out; mode 2: bias+gelu -> bf16 out; mode 3: bias+resid -> f32 out
#define BM 256
#define BN 128
#define BK 32
#define ASTR 40
#define BSTR 34

__global__ __launch_bounds__(256) void gemm_bf16(
    const unsigned short* __restrict__ A, const unsigned short* __restrict__ B,
    const float* __restrict__ bias, const float* __restrict__ resid,
    float* __restrict__ outF, unsigned short* __restrict__ outB,
    int M, int N, int K, int mode) {
    __shared__ unsigned short Ash[BM * ASTR];
    __shared__ unsigned short Bsh[BN * BSTR];   // transposed: [n][k]

    const int t = threadIdx.x;
    const int wave = t >> 5, lane = t & 31;
    const int g = lane >> 4, ln = lane & 15;
    const int wm = wave >> 1, wn = wave & 1;
    const int bm = blockIdx.y * BM, bn = blockIdx.x * BN;

    v8f acc[4][4];
#pragma unroll
    for (int i = 0; i < 4; ++i)
#pragma unroll
        for (int j = 0; j < 4; ++j) acc[i][j] = (v8f){};

    for (int k0 = 0; k0 < K; k0 += BK) {
        {   // A tile: 256x32, 1 thread/row, 32 bf16 (64B) each
            const uint4* src = (const uint4*)(A + (size_t)(bm + t) * K + k0);
            uint4 d0 = src[0], d1 = src[1], d2 = src[2], d3 = src[3];
            uint4* dst = (uint4*)(Ash + t * ASTR);
            dst[0] = d0; dst[1] = d1; dst[2] = d2; dst[3] = d3;
        }
        {   // B tile: 32x128, transpose into Bsh[n][k]
            int row = t >> 3, nseg = (t & 7) * 16;
            const uint4* src = (const uint4*)(B + (size_t)(k0 + row) * N + bn + nseg);
            uint4 d0 = src[0], d1 = src[1];
            unsigned short tmp[16];
            *(uint4*)(tmp) = d0; *(uint4*)(tmp + 8) = d1;
#pragma unroll
            for (int i = 0; i < 16; ++i) Bsh[(nseg + i) * BSTR + row] = tmp[i];
        }
        // prefetch next K-slice into cache while this one is consumed
        if (k0 + BK < K) {
            __builtin_prefetch(A + (size_t)(bm + t) * K + k0 + BK, 0, 0);
            __builtin_prefetch(B + (size_t)(k0 + BK + (t >> 3)) * N + bn + (t & 7) * 16, 0, 0);
        }
        __syncthreads();

        FragB16 af[4], bf[4];
#pragma unroll
        for (int i = 0; i < 4; ++i) {
            const unsigned short* base = Ash + (wm * 64 + i * 16 + ln) * ASTR;
#pragma unroll
            for (int p = 0; p < 8; ++p) {
                int kk = (p < 4 ? 2 * p : 16 + 2 * (p - 4)) + 8 * g;
                af[i].u[p] = *(const unsigned int*)(base + kk);
            }
        }
#pragma unroll
        for (int j = 0; j < 4; ++j) {
            const unsigned short* base = Bsh + (wn * 64 + j * 16 + ln) * BSTR + 8 * g;
#pragma unroll
            for (int p = 0; p < 8; ++p)
                bf[j].u[p] = *(const unsigned int*)(base + (p < 4 ? 2 * p : 16 + 2 * (p - 4)));
        }
#pragma unroll
        for (int i = 0; i < 4; ++i)
#pragma unroll
            for (int j = 0; j < 4; ++j)
                acc[i][j] = __builtin_amdgcn_wmma_f32_16x16x32_bf16(
                    false, af[i].v, false, bf[j].v, (short)0, acc[i][j], false, false);
        __syncthreads();
    }

#pragma unroll
    for (int i = 0; i < 4; ++i) {
#pragma unroll
        for (int j = 0; j < 4; ++j) {
            int col = bn + wn * 64 + j * 16 + ln;
            float bc = bias[col];
#pragma unroll
            for (int r = 0; r < 8; ++r) {
                int row = bm + wm * 64 + i * 16 + r + 8 * g;
                size_t idx = (size_t)row * N + col;
                float v = acc[i][j][r] + bc;
                if (mode == 3) {
                    outF[idx] = v + resid[idx];
                } else {
                    if (mode == 2) {
                        float u = 0.7978845608028654f * (v + 0.044715f * v * v * v);
                        v = 0.5f * v * (1.0f + tanhf(u));
                    }
                    outB[idx] = f2bf(v);
                }
            }
        }
    }
}

// Flash attention: S=1024, B=4, H=16, D=64. qkv bf16 [S*B, 3072], out bf16 [S*B, 1024]
__global__ __launch_bounds__(128) void attn_flash(const unsigned short* __restrict__ qkv,
                                                  unsigned short* __restrict__ out) {
    __shared__ unsigned short Qsh[64 * 72];
    __shared__ unsigned short Ksh[32 * 72];      // [key][d]
    __shared__ unsigned short Vsh[64 * 40];      // transposed: [d][key]
    __shared__ unsigned short Psh[64 * 40];      // 4 waves x 16 rows

    const int t = threadIdx.x;
    const int wave = t >> 5, lane = t & 31;
    const int g = lane >> 4, ln = lane & 15;
    const int bh = blockIdx.y, b = bh >> 4, h = bh & 15;
    const int q0 = blockIdx.x * 64;
    const int E3 = 3072;

    {   // load 64x64 Q block
        int row = t >> 1, half = t & 1;
        const uint4* src = (const uint4*)(qkv + (size_t)((q0 + row) * 4 + b) * E3 + h * 64 + half * 32);
        uint4* dst = (uint4*)(Qsh + row * 72 + half * 32);
        dst[0] = src[0]; dst[1] = src[1]; dst[2] = src[2]; dst[3] = src[3];
    }
    __syncthreads();

    FragB16 aq[2];
#pragma unroll
    for (int c = 0; c < 2; ++c) {
        const unsigned short* base = Qsh + (wave * 16 + ln) * 72 + c * 32;
#pragma unroll
        for (int p = 0; p < 8; ++p) {
            int kk = (p < 4 ? 2 * p : 16 + 2 * (p - 4)) + 8 * g;
            aq[c].u[p] = *(const unsigned int*)(base + kk);
        }
    }

    float mrow[8], lrow[8];
    v8f O[4];
#pragma unroll
    for (int r = 0; r < 8; ++r) { mrow[r] = -1.0e30f; lrow[r] = 0.f; }
#pragma unroll
    for (int j = 0; j < 4; ++j) O[j] = (v8f){};

    for (int kb = 0; kb < 1024; kb += 32) {
        __syncthreads();
        {   // K block 32x64 -> Ksh[key][d]
            int row = t >> 2, seg = (t & 3) * 16;
            const uint4* src = (const uint4*)(qkv + (size_t)((kb + row) * 4 + b) * E3 + 1024 + h * 64 + seg);
            uint4* dst = (uint4*)(Ksh + row * 72 + seg);
            dst[0] = src[0]; dst[1] = src[1];
        }
        {   // V block 32x64 -> Vsh[d][key] (transposed)
            int row = t >> 2, seg = (t & 3) * 16;
            const uint4* src = (const uint4*)(qkv + (size_t)((kb + row) * 4 + b) * E3 + 2048 + h * 64 + seg);
            uint4 d0 = src[0], d1 = src[1];
            unsigned short tmp[16];
            *(uint4*)(tmp) = d0; *(uint4*)(tmp + 8) = d1;
#pragma unroll
            for (int i = 0; i < 16; ++i) Vsh[(seg + i) * 40 + row] = tmp[i];
        }
        __syncthreads();

        // scores: 16 q rows x 32 keys
        v8f sc[2];
#pragma unroll
        for (int nt = 0; nt < 2; ++nt) {
            v8f c = (v8f){};
#pragma unroll
            for (int kc = 0; kc < 2; ++kc) {
                FragB16 bk;
                const unsigned short* base = Ksh + (nt * 16 + ln) * 72 + g * 8 + kc * 32;
#pragma unroll
                for (int p = 0; p < 8; ++p) bk.u[p] = *(const unsigned int*)(base + (p < 4 ? 2 * p : 16 + 2 * (p - 4)));
                c = __builtin_amdgcn_wmma_f32_16x16x32_bf16(false, aq[kc].v, false, bk.v,
                                                            (short)0, c, false, false);
            }
            sc[nt] = c;
        }

        // online softmax per row (rows live across 16-lane halves)
#pragma unroll
        for (int r = 0; r < 8; ++r) {
            float a0 = sc[0][r] * 0.125f, a1 = sc[1][r] * 0.125f;
            float mx = fmaxf(a0, a1);
#pragma unroll
            for (int off = 1; off < 16; off <<= 1) mx = fmaxf(mx, __shfl_xor(mx, off));
            float mnew  = fmaxf(mrow[r], mx);
            float alpha = __expf(mrow[r] - mnew);
            float p0 = __expf(a0 - mnew), p1 = __expf(a1 - mnew);
            float ps = p0 + p1;
#pragma unroll
            for (int off = 1; off < 16; off <<= 1) ps += __shfl_xor(ps, off);
            lrow[r] = lrow[r] * alpha + ps;
            mrow[r] = mnew;
#pragma unroll
            for (int j = 0; j < 4; ++j) O[j][r] *= alpha;
            unsigned short* pr = Psh + (wave * 16 + r + 8 * g) * 40;
            pr[ln] = f2bf(p0);
            pr[16 + ln] = f2bf(p1);
        }

        // P(16x32) x V(32x64) -> O  (per-wave LDS region; in-wave DS ordering)
        FragB16 ap;
        {
            const unsigned short* base = Psh + (wave * 16 + ln) * 40;
#pragma unroll
            for (int p = 0; p < 8; ++p) {
                int kk = (p < 4 ? 2 * p : 16 + 2 * (p - 4)) + 8 * g;
                ap.u[p] = *(const unsigned int*)(base + kk);
            }
        }
#pragma unroll
        for (int j = 0; j < 4; ++j) {
            FragB16 bv;
            const unsigned short* base = Vsh + (j * 16 + ln) * 40 + g * 8;
#pragma unroll
            for (int p = 0; p < 8; ++p) bv.u[p] = *(const unsigned int*)(base + (p < 4 ? 2 * p : 16 + 2 * (p - 4)));
            O[j] = __builtin_amdgcn_wmma_f32_16x16x32_bf16(false, ap.v, false, bv.v,
                                                           (short)0, O[j], false, false);
        }
    }

#pragma unroll
    for (int r = 0; r < 8; ++r) {
        float inv = 1.0f / lrow[r];
        int srow = q0 + wave * 16 + r + 8 * g;
        unsigned short* dst = out + (size_t)(srow * 4 + b) * 1024 + h * 64;
#pragma unroll
        for (int j = 0; j < 4; ++j) dst[j * 16 + ln] = f2bf(O[j][r] * inv);
    }
}

extern "C" void kernel_launch(void* const* d_in, const int* in_sizes, int n_in,
                              void* d_out, int out_size, void* d_ws, size_t ws_size,
                              hipStream_t stream) {
    (void)in_sizes; (void)n_in; (void)out_size; (void)ws_size;
    const float* x      = (const float*)d_in[0];
    const float* ln1_g  = (const float*)d_in[1];
    const float* ln1_b  = (const float*)d_in[2];
    const float* w_attn = (const float*)d_in[3];
    const float* b_attn = (const float*)d_in[4];
    const float* w_proj = (const float*)d_in[5];
    const float* b_proj = (const float*)d_in[6];
    const float* ln2_g  = (const float*)d_in[7];
    const float* ln2_b  = (const float*)d_in[8];
    const float* w_fc   = (const float*)d_in[9];
    const float* b_fc   = (const float*)d_in[10];
    const float* w_out  = (const float*)d_in[11];
    const float* b_out  = (const float*)d_in[12];

    char* ws = (char*)d_ws;
    unsigned short* wA   = (unsigned short*)(ws + 0);                 //  6 MB
    unsigned short* wP   = (unsigned short*)(ws + 6291456);           //  2 MB
    unsigned short* wF   = (unsigned short*)(ws + 8388608);           //  8 MB
    unsigned short* wO   = (unsigned short*)(ws + 16777216);          //  8 MB
    unsigned short* hbf  = (unsigned short*)(ws + 25165824);          //  8 MB (reused for LN2 out)
    unsigned short* qkvb = (unsigned short*)(ws + 33554432);          // 24 MB
    unsigned short* attn = (unsigned short*)(ws + 58720256);          //  8 MB
    unsigned short* mbf  = (unsigned short*)(ws + 33554432);          // 32 MB (overlays qkv+attn, dead by then)
    float*          r1   = (float*)(ws + 67108864);                   // 16 MB
    float* outF = (float*)d_out;

    // 1. weight conversion fp32 -> bf16
    cvt_f32_bf16<<<(3145728 + 255) / 256, 256, 0, stream>>>(w_attn, wA, 3145728);
    cvt_f32_bf16<<<(1048576 + 255) / 256, 256, 0, stream>>>(w_proj, wP, 1048576);
    cvt_f32_bf16<<<(4194304 + 255) / 256, 256, 0, stream>>>(w_fc,   wF, 4194304);
    cvt_f32_bf16<<<(4194304 + 255) / 256, 256, 0, stream>>>(w_out,  wO, 4194304);

    // 2. LN1
    ln_bf16<<<4096, 256, 0, stream>>>(x, ln1_g, ln1_b, hbf);

    // 3. qkv = h @ w_attn + b_attn  (M=4096, N=3072, K=1024) -> bf16
    gemm_bf16<<<dim3(24, 16), 256, 0, stream>>>(hbf, wA, b_attn, nullptr,
                                                nullptr, qkvb, 4096, 3072, 1024, 1);
    // 4. flash attention -> attn bf16 [S*B, E]
    attn_flash<<<dim3(16, 64), 128, 0, stream>>>(qkvb, attn);

    // 5. r1 = x + attn @ w_proj + b_proj  (M=4096, N=1024, K=1024) -> f32
    gemm_bf16<<<dim3(8, 16), 256, 0, stream>>>(attn, wP, b_proj, x,
                                               r1, nullptr, 4096, 1024, 1024, 3);
    // 6. LN2
    ln_bf16<<<4096, 256, 0, stream>>>(r1, ln2_g, ln2_b, hbf);

    // 7. m = gelu(h2 @ w_fc + b_fc)  (M=4096, N=4096, K=1024) -> bf16
    gemm_bf16<<<dim3(32, 16), 256, 0, stream>>>(hbf, wF, b_fc, nullptr,
                                                nullptr, mbf, 4096, 4096, 1024, 2);
    // 8. out = r1 + m @ w_out + b_out  (M=4096, N=1024, K=4096) -> f32
    gemm_bf16<<<dim3(8, 16), 256, 0, stream>>>(mbf, wO, b_out, r1,
                                               outF, nullptr, 4096, 1024, 4096, 3);
}
